// MambaDRnet_63582695850740
// MI455X (gfx1250) — matrix operand order repaired and checked by reading, not compiled, not run on hardware
//
#include <hip/hip_runtime.h>

// CDNA5 / gfx1250, wave32. WMMA f32_16x16x32_f16 for all three Mamba GEMMs.

typedef __attribute__((ext_vector_type(16))) _Float16 v16h;
typedef __attribute__((ext_vector_type(8)))  float    v8f;

#define SEQ   192
#define LSEQ  1000
#define DM    32
#define DI    64
#define DSTATE 16
#define TL    64   // rows per tile

__device__ __forceinline__ float sigmoidf_(float x){ return 1.f/(1.f+__expf(-x)); }
__device__ __forceinline__ float siluf_(float x){ return x*sigmoidf_(x); }
__device__ __forceinline__ float softplusf_(float x){ return (x>20.f)?x:__logf(1.f+__expf(x)); }

// A-matrix (16x32 f16) swizzle: element (m,k) -> lane, half  (contiguous k: lane/16 = k/16, half = k%16)
__device__ __forceinline__ int aswz_lane(int m,int k){ return m + ((k>>4)<<4); }
__device__ __forceinline__ int aswz_half(int k){ return k&15; }

// ---------------------------------------------------------------------------
// Pack one branch's GEMM weights (f32) into a B-matrix f16 lane layout (contiguous k, same as the A swizzle):
// B (32xN f16): lane = n%16 + 16*(k>=16), half = k%16 per 16x16 N-tile / 32-K-tile
// ---------------------------------------------------------------------------
__global__ void pack_weights(const float* __restrict__ in_w,   // 32x128
                             const float* __restrict__ xp_w,   // 64x34
                             const float* __restrict__ ow,     // 64x32
                             _Float16* __restrict__ in_pk,     // 8 tiles * 512
                             _Float16* __restrict__ xp_pk,     // 6 tiles * 512 (N padded to 48)
                             _Float16* __restrict__ out_pk) {  // 4 tiles * 512
  int tid = threadIdx.x;
  for (int i = tid; i < 8*512; i += blockDim.x) {
    int nt=i>>9, r=i&511, lane=r>>4, h=r&15;
    int k = h + ((lane>>4)<<4);
    int n = nt*16 + (lane&15);
    in_pk[i] = (_Float16)in_w[k*128 + n];
  }
  for (int i = tid; i < 6*512; i += blockDim.x) {
    int tile=i>>9, r=i&511, lane=r>>4, h=r&15;
    int nt=tile>>1, kt=tile&1;
    int k = kt*32 + h + ((lane>>4)<<4);
    int n = nt*16 + (lane&15);
    xp_pk[i] = (_Float16)((n < 34) ? xp_w[k*34 + n] : 0.f);
  }
  for (int i = tid; i < 4*512; i += blockDim.x) {
    int tile=i>>9, r=i&511, lane=r>>4, h=r&15;
    int nt=tile>>1, kt=tile&1;
    int k = kt*32 + h + ((lane>>4)<<4);
    int n = nt*16 + (lane&15);
    out_pk[i] = (_Float16)ow[k*32 + n];
  }
}

// ---------------------------------------------------------------------------
// Fused residual-Mamba branch. One block per sequence row (192 blocks),
// 256 threads = 8 waves. Tiles of TL=64 over L=1000; scan state in registers.
// ---------------------------------------------------------------------------
__global__ void __launch_bounds__(256)
mamba_branch(const float* __restrict__ x,
             const float* __restrict__ cw_in, const float* __restrict__ cb_in,
             int KW, int PAD,
             const float* __restrict__ ln_g, const float* __restrict__ ln_b,
             const _Float16* __restrict__ in_pk,
             const float* __restrict__ conv_w, const float* __restrict__ conv_b,
             const _Float16* __restrict__ xp_pk,
             const float* __restrict__ dt_w, const float* __restrict__ dt_b,
             const float* __restrict__ A_log, const float* __restrict__ Dp,
             const _Float16* __restrict__ out_pk,
             float* __restrict__ R)
{
  __shared__ float    s_t[TL][DM];      // layernorm input / residual (8KB)
  __shared__ v16h     s_tln[4][32];     // ln output, A-swizzled f16 (4KB)
  __shared__ float    s_xr[TL][DI];     // raw xr; reused as y_swz f16 (16KB)
  __shared__ v16h     s_xrs[8][32];     // silu(conv(xr)) A-swizzled (8KB)
  __shared__ _Float16 s_z[TL][DI];      // silu(z) (8KB)
  __shared__ float    s_xdbl[TL][48];   // dt(2) | B(16) | C(16) | pad (12KB)
  __shared__ float    s_red[4][DI];     // scan reduction (1KB)
  __shared__ float    s_hist[3][DI];    // xr history across tiles (0.75KB)

  const int tid  = threadIdx.x;
  const int wv   = tid >> 5;
  const int lane = tid & 31;
  const int seq  = blockIdx.x;
  const int bb   = seq / 12, hh = seq % 12;

  // scan ownership: thread owns (d = sd, states s = sg*4 .. sg*4+3)
  const int sd = tid & 63;
  const int sg = tid >> 6;
  float hstate[4] = {0.f, 0.f, 0.f, 0.f};
  float Areg[4];
  const float dw0 = dt_w[sd], dw1 = dt_w[64 + sd], dbr = dt_b[sd], Dr = Dp[sd];
  #pragma unroll
  for (int j = 0; j < 4; ++j) Areg[j] = -__expf(A_log[sd*DSTATE + sg*4 + j]);

  if (tid < 192) s_hist[tid >> 6][tid & 63] = 0.f;
  __syncthreads();

  v16h* y_swz = (v16h*)&s_xr[0][0];   // 8 x 32 v16h region aliasing s_xr

  for (int tile = 0; tile < (LSEQ + TL - 1) / TL; ++tile) {
    const int t0 = tile * TL;
    const int valid = (LSEQ - t0 < TL) ? (LSEQ - t0) : TL;

    // prefetch next tile of x rows (global_prefetch_b8)
    if (tid < 2 && t0 + TL < LSEQ)
      __builtin_prefetch(x + ((bb*2 + tid)*12 + hh)*LSEQ + t0 + TL, 0, 1);

    // (1) input conv (2 -> 32 channels, kernel KW along W) -> s_t
    for (int i = tid; i < TL*DM; i += 256) {
      int row = i >> 5, c = i & 31;
      int w = t0 + row;
      float acc = 0.f;
      if (w < LSEQ) {
        acc = cb_in[c];
        for (int cin = 0; cin < 2; ++cin) {
          const float* xrow = x + ((bb*2 + cin)*12 + hh)*LSEQ;
          const float* wrow = cw_in + (c*2 + cin)*KW;
          for (int kw = 0; kw < KW; ++kw) {
            int wi = w + kw - PAD;
            if (wi >= 0 && wi < LSEQ) acc += xrow[wi] * wrow[kw];
          }
        }
      }
      s_t[row][c] = acc;
    }
    __syncthreads();

    // layernorm per row -> f16 A-swizzled tile
    if (tid < TL) {
      int row = tid;
      float mu = 0.f;
      for (int c = 0; c < DM; ++c) mu += s_t[row][c];
      mu *= (1.f / DM);
      float var = 0.f;
      for (int c = 0; c < DM; ++c) { float d0 = s_t[row][c] - mu; var += d0*d0; }
      var *= (1.f / DM);
      float inv = rsqrtf(var + 1e-5f);
      int mt = row >> 4, m = row & 15;
      for (int c = 0; c < DM; ++c) {
        float v = (s_t[row][c] - mu) * inv * ln_g[c] + ln_b[c];
        ((_Float16*)&s_tln[mt][aswz_lane(m, c)])[aswz_half(c)] = (_Float16)v;
      }
    }
    __syncthreads();

    // (2) GEMM1: xz[64][128] = t_ln @ in_w   (K=32, 4 mt x 8 nt, 1 WMMA each).
    // Wave-group-uniform split: waves 0..3 -> xr columns, waves 4..7 -> z columns.
    {
      const int mt = wv & 3;
      const int m0 = (lane >> 4) << 3;
      const int n  = lane & 15;
      v16h a = s_tln[mt][lane];
      if (wv < 4) {            // xr half: nt = 0..3, straight stores
        #pragma unroll
        for (int nt = 0; nt < 4; ++nt) {
          v16h b = *((const v16h*)in_pk + nt*32 + lane);
          v8f c = {};
          c = __builtin_amdgcn_wmma_f32_16x16x32_f16(false, a, false, b, (short)0, c, false, false);
          #pragma unroll
          for (int r = 0; r < 8; ++r) s_xr[mt*16 + m0 + r][nt*16 + n] = c[r];
        }
      } else {                 // z half: nt = 4..7, silu + f16 stores
        #pragma unroll
        for (int j = 0; j < 4; ++j) {
          int nt = 4 + j;
          v16h b = *((const v16h*)in_pk + nt*32 + lane);
          v8f c = {};
          c = __builtin_amdgcn_wmma_f32_16x16x32_f16(false, a, false, b, (short)0, c, false, false);
          #pragma unroll
          for (int r = 0; r < 8; ++r)
            s_z[mt*16 + m0 + r][j*16 + n] = (_Float16)siluf_(c[r]);
        }
      }
    }
    __syncthreads();

    // (3) depthwise causal conv (k=4) + silu -> A-swizzled f16
    for (int i = tid; i < TL*DI; i += 256) {
      int row = i >> 6, d = i & 63;
      float acc = conv_b[d];
      #pragma unroll
      for (int kk = 0; kk < 4; ++kk) {
        int tt = row + kk - 3;
        float xv = (tt >= 0) ? s_xr[tt][d] : s_hist[tt + 3][d];
        acc += xv * conv_w[d*4 + kk];
      }
      float sl = siluf_(acc);
      int mt = row >> 4, m = row & 15, kt = d >> 5, k = d & 31;
      ((_Float16*)&s_xrs[mt*2 + kt][aswz_lane(m, k)])[aswz_half(k)] = (_Float16)sl;
    }
    __syncthreads();

    // save last 3 raw xr rows for next tile's conv
    if (tid < 192) { int j = tid >> 6, d = tid & 63; s_hist[j][d] = s_xr[61 + j][d]; }
    __syncthreads();

    // (4) GEMM2: x_dbl[64][48] = xr_silu @ xproj (K=64 -> 2 accumulating WMMAs)
    #pragma unroll
    for (int pass = 0; pass < 2; ++pass) {
      int tileid = wv + pass*8;
      if (tileid < 12) {
        int mt = tileid & 3, nt = tileid >> 2;
        v8f c = {};
        #pragma unroll
        for (int kt = 0; kt < 2; ++kt) {
          v16h a = s_xrs[mt*2 + kt][lane];
          v16h b = *((const v16h*)xp_pk + (nt*2 + kt)*32 + lane);
          c = __builtin_amdgcn_wmma_f32_16x16x32_f16(false, a, false, b, (short)0, c, false, false);
        }
        int m0 = (lane >> 4) << 3, n = lane & 15;
        #pragma unroll
        for (int r = 0; r < 8; ++r) s_xdbl[mt*16 + m0 + r][nt*16 + n] = c[r];
      }
    }
    __syncthreads();

    // (5) selective scan (sequential in t), fp32; y -> A-swizzled f16 (alias s_xr)
    for (int t = 0; t < valid; ++t) {
      float dt0 = s_xdbl[t][0], dt1 = s_xdbl[t][1];
      float delta = softplusf_(dt0*dw0 + dt1*dw1 + dbr);
      int mtu = t >> 4, mlu = t & 15, ktu = sd >> 5, ku = sd & 31;
      float u = (float)((_Float16*)&s_xrs[mtu*2 + ktu][aswz_lane(mlu, ku)])[aswz_half(ku)];
      float du = delta * u;
      float ps = 0.f;
      #pragma unroll
      for (int j = 0; j < 4; ++j) {
        int s = sg*4 + j;
        float Bv = s_xdbl[t][2 + s];
        float Cv = s_xdbl[t][18 + s];
        hstate[j] = __expf(delta * Areg[j]) * hstate[j] + du * Bv;
        ps += hstate[j] * Cv;
      }
      s_red[sg][sd] = ps;
      __syncthreads();
      if (sg == 0) {
        float y = s_red[0][sd] + s_red[1][sd] + s_red[2][sd] + s_red[3][sd];
        y = (y + u * Dr) * (float)s_z[t][sd];
        ((_Float16*)&y_swz[(mtu*2 + ktu)*32 + aswz_lane(mlu, ku)])[aswz_half(ku)] = (_Float16)y;
      }
      __syncthreads();
    }

    // (6) GEMM3: out[64][32] = y @ out_w (K=64), + residual, store
    {
      int mt = wv & 3, nt = wv >> 2;   // 8 tiles, one per wave
      v8f c = {};
      #pragma unroll
      for (int kt = 0; kt < 2; ++kt) {
        v16h a = y_swz[(mt*2 + kt)*32 + lane];
        v16h b = *((const v16h*)out_pk + (nt*2 + kt)*32 + lane);
        c = __builtin_amdgcn_wmma_f32_16x16x32_f16(false, a, false, b, (short)0, c, false, false);
      }
      int m0 = (lane >> 4) << 3, n = lane & 15;
      #pragma unroll
      for (int r = 0; r < 8; ++r) {
        int mrow = mt*16 + m0 + r;
        if (mrow < valid) {
          float vout = c[r] + s_t[mrow][nt*16 + n];
          R[(size_t)(seq*LSEQ + t0 + mrow)*DM + nt*16 + n] = vout;
        }
      }
    }
    __syncthreads();
  }
}

// ---------------------------------------------------------------------------
// Final 32->1 convs on both branches, summed (+ both biases)
// ---------------------------------------------------------------------------
__global__ void final_conv(const float* __restrict__ R2, const float* __restrict__ R4,
                           const float* __restrict__ c12_w, const float* __restrict__ c12_b,
                           const float* __restrict__ c14_w, const float* __restrict__ c14_b,
                           float* __restrict__ out)
{
  int idx = blockIdx.x * blockDim.x + threadIdx.x;
  if (idx >= 16*12*1000) return;
  int w = idx % 1000;
  int h = (idx / 1000) % 12;
  int b = idx / 12000;
  int seq = b*12 + h;
  const float* r2 = R2 + (size_t)seq*LSEQ*DM;
  const float* r4 = R4 + (size_t)seq*LSEQ*DM;
  float acc = c12_b[0] + c14_b[0];
  for (int kw = 0; kw < 5; ++kw) {
    int wi = w + kw - 2;
    if (wi >= 0 && wi < LSEQ) {
      const float* p = r2 + wi*DM;
      for (int c = 0; c < DM; ++c) acc += p[c] * c12_w[c*5 + kw];
    }
  }
  for (int kw = 0; kw < 15; ++kw) {
    int wi = w + kw - 7;
    if (wi >= 0 && wi < LSEQ) {
      const float* p = r4 + wi*DM;
      for (int c = 0; c < DM; ++c) acc += p[c] * c14_w[c*15 + kw];
    }
  }
  out[idx] = acc;
}

// ---------------------------------------------------------------------------
extern "C" void kernel_launch(void* const* d_in, const int* in_sizes, int n_in,
                              void* d_out, int out_size, void* d_ws, size_t ws_size,
                              hipStream_t stream)
{
  (void)in_sizes; (void)n_in; (void)out_size; (void)ws_size;
  const float* x     = (const float*)d_in[0];
  const float* c21_w = (const float*)d_in[1];
  const float* c21_b = (const float*)d_in[2];
  const float* c41_w = (const float*)d_in[3];
  const float* c41_b = (const float*)d_in[4];
  const float* c12_w = (const float*)d_in[5];
  const float* c12_b = (const float*)d_in[6];
  const float* c14_w = (const float*)d_in[7];
  const float* c14_b = (const float*)d_in[8];

  // workspace layout
  float* R2 = (float*)d_ws;                       // 192*1000*32 f32
  float* R4 = R2 + (size_t)SEQ*LSEQ*DM;           // 192*1000*32 f32
  _Float16* pk = (_Float16*)(R4 + (size_t)SEQ*LSEQ*DM);
  _Float16* m2_in_pk  = pk;                // 4096 halfs
  _Float16* m2_xp_pk  = pk + 4096;         // 3072
  _Float16* m2_out_pk = pk + 4096 + 3072;  // 2048
  _Float16* m4_in_pk  = pk + 9216;
  _Float16* m4_xp_pk  = pk + 9216 + 4096;
  _Float16* m4_out_pk = pk + 9216 + 4096 + 3072;

  pack_weights<<<1, 256, 0, stream>>>((const float*)d_in[11], (const float*)d_in[14],
                                      (const float*)d_in[19],
                                      m2_in_pk, m2_xp_pk, m2_out_pk);
  pack_weights<<<1, 256, 0, stream>>>((const float*)d_in[22], (const float*)d_in[25],
                                      (const float*)d_in[30],
                                      m4_in_pk, m4_xp_pk, m4_out_pk);

  mamba_branch<<<SEQ, 256, 0, stream>>>(x, c21_w, c21_b, 5, 2,
      (const float*)d_in[9],  (const float*)d_in[10], m2_in_pk,
      (const float*)d_in[12], (const float*)d_in[13], m2_xp_pk,
      (const float*)d_in[15], (const float*)d_in[16],
      (const float*)d_in[17], (const float*)d_in[18], m2_out_pk, R2);

  mamba_branch<<<SEQ, 256, 0, stream>>>(x, c41_w, c41_b, 15, 7,
      (const float*)d_in[20], (const float*)d_in[21], m4_in_pk,
      (const float*)d_in[23], (const float*)d_in[24], m4_xp_pk,
      (const float*)d_in[26], (const float*)d_in[27],
      (const float*)d_in[28], (const float*)d_in[29], m4_out_pk, R4);

  final_conv<<<(16*12*1000 + 255) / 256, 256, 0, stream>>>(
      R2, R4, c12_w, c12_b, c14_w, c14_b, (float*)d_out);
}
